// GuideHead_70368744178364
// MI455X (gfx1250) — matrix operand order repaired and checked by reading, not compiled, not run on hardware
//
#include <hip/hip_runtime.h>

#define S_ 1024
#define D_ 1024
#define B_ 32
#define LDSTR 40   // ushort stride per LDS row (32 data + 8 pad) -> 80B = 20 banks

// 1 = gfx1250 GLOBAL_LOAD_ASYNC_TO_LDS_B128 + ping-pong LDS double buffering.
// 0 = batched VGPR-mediated copy, single-stage.
#define USE_ASYNC_COPY 1

typedef __attribute__((ext_vector_type(16))) __bf16 v16bf;
typedef __attribute__((ext_vector_type(8)))  float  v8f;

// ---------- helpers ----------

// Split fp32 into hi/lo bf16 (both RNE). x ~= hi + lo to ~16 mantissa bits.
__device__ __forceinline__ void splitbf(float x, unsigned short& h, unsigned short& l) {
    unsigned u  = __float_as_uint(x);
    unsigned rh = u + 0x7FFFu + ((u >> 16) & 1u);
    h = (unsigned short)(rh >> 16);
    float fh = __uint_as_float(((unsigned)h) << 16);
    float r  = x - fh;
    unsigned v  = __float_as_uint(r);
    unsigned rl = v + 0x7FFFu + ((v >> 16) & 1u);
    l = (unsigned short)(rl >> 16);
}

__device__ __forceinline__ void cvt_store4(unsigned short* hi, unsigned short* lo, float4 f) {
    unsigned short h0,h1,h2,h3,l0,l1,l2,l3;
    splitbf(f.x,h0,l0); splitbf(f.y,h1,l1); splitbf(f.z,h2,l2); splitbf(f.w,h3,l3);
    ((unsigned*)hi)[0] = (unsigned)h0 | ((unsigned)h1 << 16);
    ((unsigned*)hi)[1] = (unsigned)h2 | ((unsigned)h3 << 16);
    ((unsigned*)lo)[0] = (unsigned)l0 | ((unsigned)l1 << 16);
    ((unsigned*)lo)[1] = (unsigned)l2 | ((unsigned)l3 << 16);
}

#if USE_ASYNC_COPY
// Async DMA: 16B per lane, global -> LDS, no VGPR round-trip. ASYNCcnt-tracked.
__device__ __forceinline__ void async_copy16(const unsigned short* gbase, unsigned voff,
                                             unsigned short* ldst) {
    asm volatile("global_load_async_to_lds_b128 %0, %1, %2"
                 :: "v"((unsigned)(unsigned long long)ldst), "v"(voff), "s"(gbase)
                 : "memory");
}
#endif

// A fragment (16x32 bf16): lane holds K = kb..kb+7 and kb+16..kb+23 (kb = 8*(lane/16))
__device__ __forceinline__ v16bf load_fragA(const unsigned short* p) {
    union { uint4 q[2]; v16bf v; } u;
    u.q[0] = *(const uint4*)(p);
    u.q[1] = *(const uint4*)(p + 16);
    return u.v;
}
// B fragment (32x16 bf16): lane = column, K = kb..kb+7 and kb+16..kb+23 (kb = 8*(lane/16))
__device__ __forceinline__ v16bf load_fragB(const unsigned short* p) {
    union { uint4 q[2]; v16bf v; } u;
    u.q[0] = *(const uint4*)(p);
    u.q[1] = *(const uint4*)(p + 16);
    return u.v;
}

// ---------- kernel 0: one-time fp32 -> split-bf16 plane conversion ----------
__global__ __launch_bounds__(256) void conv_kernel(
    const float* __restrict__ src, const float* __restrict__ tgt,
    unsigned short* __restrict__ srcHi, unsigned short* __restrict__ srcLo,
    unsigned short* __restrict__ tgtHi, unsigned short* __restrict__ tgtLo)
{
    const int mat = blockIdx.y;
    const float* in = mat ? tgt : src;
    unsigned short* hp = mat ? tgtHi : srcHi;
    unsigned short* lp = mat ? tgtLo : srcLo;
    size_t base = (size_t)blockIdx.x * 1024 + threadIdx.x;
#pragma unroll
    for (int j = 0; j < 4; j++) {
        size_t i4 = base + (size_t)j * 256;
        float4 f = ((const float4*)in)[i4];
        cvt_store4(hp + i4 * 4, lp + i4 * 4, f);
    }
}

// ---------- kernel 1: masks + lengths ----------
__global__ __launch_bounds__(256) void prep_kernel(
    const int* __restrict__ ids_src, const int* __restrict__ ids_tgt,
    float* __restrict__ bias0, float* __restrict__ bias1,
    float* __restrict__ lenArr)
{
    int b = blockIdx.x;
    __shared__ float redS[256], redT[256];
    float cs = 0.f, ct = 0.f;
    for (int i = threadIdx.x; i < S_; i += 256) {
        int is = ids_src[b * S_ + i];
        int it = ids_tgt[b * S_ + i];
        bool ms = (is == 0) | (is == 101) | (is == 102);
        bool mt = (it == 0) | (it == 101) | (it == 102);
        bias1[b * S_ + i] = ms ? -10000.f : 0.f;
        bias0[b * S_ + i] = mt ? -10000.f : 0.f;
        cs += ms ? 0.f : 1.f;
        ct += mt ? 0.f : 1.f;
    }
    redS[threadIdx.x] = cs; redT[threadIdx.x] = ct;
    __syncthreads();
    for (int off = 128; off > 0; off >>= 1) {
        if (threadIdx.x < off) {
            redS[threadIdx.x] += redS[threadIdx.x + off];
            redT[threadIdx.x] += redT[threadIdx.x + off];
        }
        __syncthreads();
    }
    if (threadIdx.x == 0) {
        lenArr[b]      = redS[0];   // len_src
        lenArr[B_ + b] = redT[0];   // len_tgt
    }
}

// ---------- kernel 2: fused split-bf16 WMMA GEMM + flash softmax-guide reduce ----------
template <int PRE>
__global__ __launch_bounds__(256) void guide_flash_kernel(
    const float* __restrict__ src, const float* __restrict__ tgt,
    const unsigned short* __restrict__ srcHi, const unsigned short* __restrict__ srcLo,
    const unsigned short* __restrict__ tgtHi, const unsigned short* __restrict__ tgtLo,
    const float* __restrict__ guide, const float* __restrict__ biasAll,
    float* __restrict__ partial /* [2][B][8 rb][8 w][2 half] */)
{
    const int rb = blockIdx.x, b = blockIdx.y, dir = blockIdx.z;
    const float* bias = biasAll + (size_t)dir * (B_ * S_) + (size_t)b * S_;
    const int rowbase = rb * 128;

    // ping-pong buffers (80 KB total; WGP LDS is 320 KB)
    __shared__ unsigned short aHi[2][128 * LDSTR], aLo[2][128 * LDSTR];
    __shared__ unsigned short bHi[2][128 * LDSTR], bLo[2][128 * LDSTR];

    const int tid = threadIdx.x, lane = tid & 31, w = tid >> 5;
    const int rowLocal = 16 * w + (lane & 15);
    const int kbA = (lane >> 4) * 8;
    const int kbB = (lane >> 4) * 8;

    const float* Af = ((dir == 0) ? src : tgt) + (size_t)b * S_ * D_ + (size_t)rowbase * D_;
    const float* Bf = ((dir == 0) ? tgt : src) + (size_t)b * S_ * D_;
    const size_t mb = (size_t)b * S_ * D_;
    const unsigned short* AHb = ((dir == 0) ? srcHi : tgtHi) + mb + (size_t)rowbase * D_;
    const unsigned short* ALb = ((dir == 0) ? srcLo : tgtLo) + mb + (size_t)rowbase * D_;
    const unsigned short* BHb = ((dir == 0) ? tgtHi : srcHi) + mb;
    const unsigned short* BLb = ((dir == 0) ? tgtLo : srcLo) + mb;
    const float* gbase = guide + (size_t)b * S_ * S_;

    float m[8], lsum[8], gsum[8];
#pragma unroll
    for (int v = 0; v < 8; v++) { m[v] = -3.0e38f; lsum[v] = 0.f; gsum[v] = 0.f; }

    for (int nb = 0; nb < 8; nb++) {
        const int colbase = nb * 128;
        v8f C[8];
        const v8f zero = {0.f,0.f,0.f,0.f,0.f,0.f,0.f,0.f};
#pragma unroll
        for (int i = 0; i < 8; i++) C[i] = zero;

#if USE_ASYNC_COPY
        // issue 8 async b128 loads for chunk kcc into buffer pbuf
        auto issue_chunk = [&](int kcc, int pbuf) {
#pragma unroll
            for (int j = 0; j < 2; j++) {
                int idx = tid + 256 * j;   // 0..511 -> 128 rows x 4 uint4
                int row = idx >> 2;
                int k8  = (idx & 3) * 8;
                unsigned offA = (unsigned)(((unsigned)row * D_ + kcc * 32 + k8) * 2);
                unsigned offB = (unsigned)(((unsigned)(colbase + row) * D_ + kcc * 32 + k8) * 2);
                async_copy16(AHb, offA, &aHi[pbuf][row * LDSTR + k8]);
                async_copy16(ALb, offA, &aLo[pbuf][row * LDSTR + k8]);
                async_copy16(BHb, offB, &bHi[pbuf][row * LDSTR + k8]);
                async_copy16(BLb, offB, &bLo[pbuf][row * LDSTR + k8]);
            }
        };
        if (PRE) issue_chunk(0, 0);   // prologue: fill buffer 0 (safe: no readers)
#endif

        for (int kc = 0; kc < 32; kc++) {      // K chunks of 32 over D=1024
            const int p = kc & 1;
            __syncthreads();                   // all waves done computing chunk kc-1
            if (PRE) {
#if USE_ASYNC_COPY
                if (kc < 31) {
                    issue_chunk(kc + 1, p ^ 1);            // prefetch next chunk
                    // in-order completion: <=8 outstanding means chunk kc's 8 are done
                    asm volatile("s_wait_asynccnt 0x8" ::: "memory");
                } else {
                    asm volatile("s_wait_asynccnt 0x0" ::: "memory");
                }
#else
                // batched copy: 8 loads into distinct temps -> stores
                uint4 t[8];
                int rows[2], k8s[2];
#pragma unroll
                for (int j = 0; j < 2; j++) {
                    int idx = tid + 256 * j;
                    rows[j] = idx >> 2;
                    k8s[j]  = (idx & 3) * 8;
                    size_t go  = (size_t)rows[j] * D_ + kc * 32 + k8s[j];
                    size_t gb2 = (size_t)(colbase + rows[j]) * D_ + kc * 32 + k8s[j];
                    t[4*j+0] = *(const uint4*)(AHb + go);
                    t[4*j+1] = *(const uint4*)(ALb + go);
                    t[4*j+2] = *(const uint4*)(BHb + gb2);
                    t[4*j+3] = *(const uint4*)(BLb + gb2);
                }
#pragma unroll
                for (int j = 0; j < 2; j++) {
                    *(uint4*)&aHi[p][rows[j] * LDSTR + k8s[j]] = t[4*j+0];
                    *(uint4*)&aLo[p][rows[j] * LDSTR + k8s[j]] = t[4*j+1];
                    *(uint4*)&bHi[p][rows[j] * LDSTR + k8s[j]] = t[4*j+2];
                    *(uint4*)&bLo[p][rows[j] * LDSTR + k8s[j]] = t[4*j+3];
                }
#endif
            } else {
                // fallback: load fp32 + split-convert on the fly
#pragma unroll
                for (int j = 0; j < 4; j++) {
                    int idx = tid + 256 * j;   // 0..1023 -> 128 rows x 8 float4
                    int row = idx >> 3;
                    int k4  = (idx & 7) << 2;
                    float4 fa = *(const float4*)(Af + (size_t)row * D_ + kc * 32 + k4);
                    cvt_store4(&aHi[p][row * LDSTR + k4], &aLo[p][row * LDSTR + k4], fa);
                    float4 fb = *(const float4*)(Bf + (size_t)(colbase + row) * D_ + kc * 32 + k4);
                    cvt_store4(&bHi[p][row * LDSTR + k4], &bLo[p][row * LDSTR + k4], fb);
                }
            }
            __syncthreads();                   // chunk kc data visible to all waves

            v16bf ah = load_fragA(&aHi[p][rowLocal * LDSTR + kbA]);
            v16bf al = load_fragA(&aLo[p][rowLocal * LDSTR + kbA]);
#pragma unroll
            for (int ct = 0; ct < 8; ct++) {
                int colL = ct * 16 + (lane & 15);
                v16bf bh = load_fragB(&bHi[p][colL * LDSTR + kbB]);
                v16bf bl = load_fragB(&bLo[p][colL * LDSTR + kbB]);
                // 3-term split: hi*hi + hi*lo + lo*hi  (lo*lo ~ 2^-16 rel, dropped)
                C[ct] = __builtin_amdgcn_wmma_f32_16x16x32_bf16(false, ah, false, bh, (short)0, C[ct], false, false);
                C[ct] = __builtin_amdgcn_wmma_f32_16x16x32_bf16(false, ah, false, bl, (short)0, C[ct], false, false);
                C[ct] = __builtin_amdgcn_wmma_f32_16x16x32_bf16(false, al, false, bh, (short)0, C[ct], false, false);
            }
        }

        // key-side bias, then per-row online softmax update (rows live in 16-lane halves)
#pragma unroll
        for (int ct = 0; ct < 8; ct++) {
            float bb = bias[colbase + ct * 16 + (lane & 15)];
#pragma unroll
            for (int e = 0; e < 8; e++) C[ct][e] += bb;
        }
#pragma unroll
        for (int v = 0; v < 8; v++) {
            float mt = -3.0e38f;
#pragma unroll
            for (int ct = 0; ct < 8; ct++) mt = fmaxf(mt, C[ct][v]);
            mt = fmaxf(mt, __shfl_xor(mt, 1));
            mt = fmaxf(mt, __shfl_xor(mt, 2));
            mt = fmaxf(mt, __shfl_xor(mt, 4));
            mt = fmaxf(mt, __shfl_xor(mt, 8));
            float mn = fmaxf(m[v], mt);
            float sc = __expf(m[v] - mn);
            lsum[v] *= sc; gsum[v] *= sc; m[v] = mn;

            int rowG = rowbase + 16 * w + v + 8 * (lane >> 4);
            float le = 0.f, ge = 0.f;
#pragma unroll
            for (int ct = 0; ct < 8; ct++) {
                int colG = colbase + ct * 16 + (lane & 15);
                float e  = __expf(C[ct][v] - mn);
                size_t gidx = (dir == 0) ? ((size_t)rowG * S_ + colG)
                                         : ((size_t)colG * S_ + rowG);
                float gv = gbase[gidx];
                le += e;
                ge += e * gv;
            }
            le += __shfl_xor(le, 1); le += __shfl_xor(le, 2);
            le += __shfl_xor(le, 4); le += __shfl_xor(le, 8);
            ge += __shfl_xor(ge, 1); ge += __shfl_xor(ge, 2);
            ge += __shfl_xor(ge, 4); ge += __shfl_xor(ge, 8);
            lsum[v] += le; gsum[v] += ge;
        }
    }

    float acc = 0.f;
#pragma unroll
    for (int v = 0; v < 8; v++) acc += gsum[v] / lsum[v];
    if ((lane & 15) == 0) {
        int idx = (((dir * B_ + b) * 8 + rb) * 8 + w) * 2 + (lane >> 4);
        partial[idx] = acc;   // unique slot -> deterministic
    }
}

// ---------- kernel 3: deterministic final reduction ----------
__global__ __launch_bounds__(256) void finish_kernel(
    const float* __restrict__ partial, const float* __restrict__ lenArr,
    float* __restrict__ out)
{
    __shared__ float sacc[256];
    int p = threadIdx.x >> 2, sub = threadIdx.x & 3;
    float s = 0.f;
    for (int i = sub; i < 128; i += 4) s += partial[p * 128 + i];
    sacc[threadIdx.x] = s;
    __syncthreads();
    if (sub == 0) sacc[threadIdx.x] = sacc[threadIdx.x] + sacc[threadIdx.x + 1]
                                    + sacc[threadIdx.x + 2] + sacc[threadIdx.x + 3];
    __syncthreads();
    if (threadIdx.x == 0) {
        float loss = 0.f;
        for (int b = 0; b < B_; b++) {
            loss += sacc[4 * b]        / lenArr[b];        // so_src / len_src
            loss += sacc[4 * (B_ + b)] / lenArr[B_ + b];   // so_tgt / len_tgt
        }
        out[0] = -loss / (float)B_;
    }
}

// ---------- launcher ----------
extern "C" void kernel_launch(void* const* d_in, const int* in_sizes, int n_in,
                              void* d_out, int out_size, void* d_ws, size_t ws_size,
                              hipStream_t stream)
{
    const float* src     = (const float*)d_in[0];
    const float* tgt     = (const float*)d_in[1];
    const int*   ids_src = (const int*)d_in[2];
    const int*   ids_tgt = (const int*)d_in[3];
    const float* guide   = (const float*)d_in[4];

    const size_t planeElems = (size_t)B_ * S_ * D_;
    const size_t planeBytes = planeElems * sizeof(unsigned short); // 64 MB
    const size_t tailFloats = (size_t)2 * B_ * S_ + 64 + 8192;
    const bool   preconv    = ws_size >= 4 * planeBytes + tailFloats * sizeof(float);

    char* wsb = (char*)d_ws;
    unsigned short* srcHi = nullptr; unsigned short* srcLo = nullptr;
    unsigned short* tgtHi = nullptr; unsigned short* tgtLo = nullptr;
    float* tail;
    if (preconv) {
        srcHi = (unsigned short*)(wsb);
        srcLo = (unsigned short*)(wsb + 1 * planeBytes);
        tgtHi = (unsigned short*)(wsb + 2 * planeBytes);
        tgtLo = (unsigned short*)(wsb + 3 * planeBytes);
        tail  = (float*)(wsb + 4 * planeBytes);
    } else {
        tail  = (float*)wsb;
    }
    float* bias0   = tail;
    float* bias1   = tail + (size_t)B_ * S_;
    float* lenArr  = tail + (size_t)2 * B_ * S_;
    float* partial = lenArr + 64;

    prep_kernel<<<dim3(B_), dim3(256), 0, stream>>>(ids_src, ids_tgt, bias0, bias1, lenArr);
    if (preconv) {
        conv_kernel<<<dim3(8192, 2), dim3(256), 0, stream>>>(src, tgt, srcHi, srcLo, tgtHi, tgtLo);
        guide_flash_kernel<1><<<dim3(8, B_, 2), dim3(256), 0, stream>>>(
            src, tgt, srcHi, srcLo, tgtHi, tgtLo, guide, bias0, partial);
    } else {
        guide_flash_kernel<0><<<dim3(8, B_, 2), dim3(256), 0, stream>>>(
            src, tgt, srcHi, srcLo, tgtHi, tgtLo, guide, bias0, partial);
    }
    finish_kernel<<<dim3(1), dim3(256), 0, stream>>>(partial, lenArr, (float*)d_out);
}
